// TripletLoss_8615704396051
// MI455X (gfx1250) — matrix operand order repaired and checked by reading, not compiled, not run on hardware
//
#include <hip/hip_runtime.h>
#include <hip/hip_bf16.h>

// ---------------------------------------------------------------------------
// Triplet loss (batch-hard) for N=8192, D=128, C=512 on gfx1250.
// Fused f16-WMMA Gram + hardest-positive/negative mining.
//  - B tiles staged to LDS via gfx1250 async copy (global_load_async_to_lds),
//    double-buffered and overlapped with WMMA via ASYNCcnt.
//  - Mining is branchless in similarity space; min/max via inline asm
//    v_min_num/v_max_num (values provably non-NaN, no canonicalize needed).
// ---------------------------------------------------------------------------

typedef _Float16 v16h __attribute__((ext_vector_type(16)));
typedef _Float16 v8h  __attribute__((ext_vector_type(8)));
typedef _Float16 v4h  __attribute__((ext_vector_type(4)));
typedef float    v8f  __attribute__((ext_vector_type(8)));

#define N_EMB 8192
#define D_EMB 128
#define N_CLS 512
#define MARGIN 0.3f
#define COLS_PER_SLICE 1024   // 8 column slices over j
#define ROWTILES_PER_BLK 8    // 8 waves/block, one 16-row tile per wave
#define TILES_PER_SLICE (COLS_PER_SLICE / 16)

union AFrag { v16h v; v8h h[2]; };

// Single-instruction float min/max (no NaN canonicalization; inputs finite).
__device__ __forceinline__ float vmin(float a, float b) {
    float r; asm("v_min_num_f32 %0, %1, %2" : "=v"(r) : "v"(a), "v"(b)); return r;
}
__device__ __forceinline__ float vmax(float a, float b) {
    float r; asm("v_max_num_f32 %0, %1, %2" : "=v"(r) : "v"(a), "v"(b)); return r;
}

// gfx1250 async copy: 16 bytes per lane, global -> LDS, tracked by ASYNCcnt.
__device__ __forceinline__ void async_copy_b128(unsigned ldsDst, const void* gSrc) {
    asm volatile("global_load_async_to_lds_b128 %0, %1, off"
                 :: "v"(ldsDst), "v"((unsigned long long)(uintptr_t)gSrc)
                 : "memory");
}
__device__ __forceinline__ void wait_async0() {
    asm volatile("s_wait_asynccnt 0x0" ::: "memory");
}

// --- init workspace accumulators -------------------------------------------
__global__ __launch_bounds__(256) void tl_init(unsigned* __restrict__ hpEnc,
                                               unsigned* __restrict__ hnEnc,
                                               int* __restrict__ cc,
                                               float* __restrict__ lossAcc,
                                               int* __restrict__ cntAcc) {
    int i = blockIdx.x * blockDim.x + threadIdx.x;
    if (i < N_EMB) {
        hpEnc[i] = 0u;                 // encodes 0.0f (no positive seen yet)
        hnEnc[i] = 0x40400000u;        // encodes 3.0f sentinel
    }
    if (i < N_CLS) cc[i] = 0;
    if (i == 0) { *lossAcc = 0.0f; *cntAcc = 0; }
}

// --- L2-normalize rows, convert to f16 (one wave per row) ------------------
__global__ __launch_bounds__(256) void tl_normalize(const float* __restrict__ x,
                                                    _Float16* __restrict__ eh) {
    const int row  = blockIdx.x * (blockDim.x >> 5) + (threadIdx.x >> 5);
    const int lane = threadIdx.x & 31;
    const float4 f = *(const float4*)(x + (size_t)row * D_EMB + lane * 4);
    float ss = f.x * f.x + f.y * f.y + f.z * f.z + f.w * f.w;
    #pragma unroll
    for (int o = 16; o >= 1; o >>= 1) ss += __shfl_xor(ss, o, 32);
    const float inv = 1.0f / fmaxf(sqrtf(ss), 1e-12f);
    v4h out;
    out.x = (_Float16)(f.x * inv);
    out.y = (_Float16)(f.y * inv);
    out.z = (_Float16)(f.z * inv);
    out.w = (_Float16)(f.w * inv);
    *(v4h*)(eh + (size_t)row * D_EMB + lane * 4) = out;
}

// --- per-class histogram (exact validity: cnt>=2 => pos, cnt<N => neg) -----
__global__ __launch_bounds__(256) void tl_hist(const int* __restrict__ labels,
                                               int* __restrict__ cc) {
    int i = blockIdx.x * blockDim.x + threadIdx.x;
    if (i < N_EMB) atomicAdd(&cc[labels[i]], 1);
}

// --- fused WMMA Gram + hardest pos/neg mining ------------------------------
// grid: (64 row-groups, 8 column slices), block: 256 (8 waves).
// Each wave owns one 16-row tile; the block cooperatively streams 16-column
// B tiles (4KB, contiguous in row-major E) through a double-buffered LDS
// stage filled with async copies (1 global_load_async_to_lds_b128 per wave
// per tile), overlapped with the previous tile's WMMA + mining.
//
// Mining in SIMILARITY space (branchless):
//   hardest positive  <=>  MIN sim over same-label columns
//   hardest negative  <=>  MAX sim over diff-label columns
// The diagonal (j==i) is deliberately NOT excluded: sim_ii = ||e_i||^2 ~= 1
// is the maximum similarity for L2-normalized data, so it can never win the
// positive MIN when a real positive exists; rows without a real positive are
// excluded exactly by the class-count histogram.
__global__ __launch_bounds__(256) void tl_mine(const _Float16* __restrict__ Eh,
                                               const int* __restrict__ labels,
                                               unsigned* __restrict__ hpEnc,
                                               unsigned* __restrict__ hnEnc) {
    __shared__ __align__(32) _Float16 bbuf[2][16 * D_EMB];   // 2 x 4 KB

    const int wave = threadIdx.x >> 5;
    const int lane = threadIdx.x & 31;
    const int hi   = lane >> 4;     // half-wave select
    const int ln16 = lane & 15;
    const int i0    = (blockIdx.x * ROWTILES_PER_BLK + wave) * 16;
    const int jbase = blockIdx.y * COLS_PER_SLICE;

    // A fragments (16x32 f16 each), loop-invariant over columns.
    // lane<16: K {k0..k0+7, k0+16..k0+23}; lane>=16: +8 on both halves.
    v16h a[4];
    {
        const _Float16* ap = Eh + (size_t)(i0 + ln16) * D_EMB + hi * 8;
        #pragma unroll
        for (int kb = 0; kb < 4; ++kb) {
            AFrag t;
            t.h[0] = *(const v8h*)(ap + kb * 32);
            t.h[1] = *(const v8h*)(ap + kb * 32 + 16);
            a[kb] = t.v;
        }
    }

    // Labels of this lane's 8 accumulator rows (M = r + 8*hi).
    int li[8];
    #pragma unroll
    for (int r = 0; r < 8; ++r) li[r] = labels[i0 + hi * 8 + r];

    // Running min-sim over positives / max-sim over negatives.
    float hps[8], hns[8];
    #pragma unroll
    for (int r = 0; r < 8; ++r) { hps[r] = 2.0f; hns[r] = -2.0f; }

    // Async staging: wave w copies columns {2w,2w+1} of the 16-column tile
    // (tile = 4KB contiguous in E); lane covers 16B at w*512 + lane*16.
    const unsigned ldsBase[2] = { (unsigned)(uintptr_t)&bbuf[0][0],
                                  (unsigned)(uintptr_t)&bbuf[1][0] };
    const unsigned stageOff   = (unsigned)(wave * 512 + lane * 16);
    const char*    gTileBase  = (const char*)Eh;

    // Prologue: stage tile 0 into buffer 0.
    async_copy_b128(ldsBase[0] + stageOff,
                    gTileBase + (size_t)jbase * 256 + wave * 512 + lane * 16);
    wait_async0();
    __syncthreads();

    for (int t = 0; t < TILES_PER_SLICE; ++t) {
        const int j0 = jbase + t * 16;

        // Stage next tile into the other buffer (free since the barrier that
        // ended iteration t-1: everyone finished reading it).
        if (t + 1 < TILES_PER_SLICE) {
            async_copy_b128(ldsBase[(t + 1) & 1] + stageOff,
                            gTileBase + (size_t)(j0 + 16) * 256 + wave * 512 + lane * 16);
        }

        // B fragment from LDS: lane n holds column j0+n, K {k0+8h..+7,
        // k0+16+8h..+7} per half-wave h (matches A).
        const _Float16* bp = &bbuf[t & 1][0] + ln16 * D_EMB + hi * 8;
        v8f c = {};
        #pragma unroll
        for (int kb = 0; kb < 4; ++kb) {
            AFrag bt;
            bt.h[0] = *(const v8h*)(bp + kb * 32);
            bt.h[1] = *(const v8h*)(bp + kb * 32 + 16);
            v16h b = bt.v;
            c = __builtin_amdgcn_wmma_f32_16x16x32_f16(
                    false, a[kb], false, b, (short)0, c, false, false);
        }

        // Branchless mining: 1 cmp + 2 cndmask + min + max per row.
        const int lj = labels[j0 + ln16];
        #pragma unroll
        for (int r = 0; r < 8; ++r) {
            const bool  same = (lj == li[r]);
            const float sim  = c[r];
            const float pc   = same ?  sim  : 2.0f;    // positive candidate
            const float nc   = same ? -2.0f : sim;     // negative candidate
            hps[r] = vmin(hps[r], pc);
            hns[r] = vmax(hns[r], nc);
        }

        // Next tile's async stage must be resident before anyone reads it.
        wait_async0();
        __syncthreads();
    }

    // Reduce across the 16 lanes (N dimension) of each half-wave.
    #pragma unroll
    for (int r = 0; r < 8; ++r) {
        #pragma unroll
        for (int off = 8; off >= 1; off >>= 1) {
            hps[r] = vmin(hps[r], __shfl_xor(hps[r], off, 32));
            hns[r] = vmax(hns[r], __shfl_xor(hns[r], off, 32));
        }
    }

    // Convert to distances once per slice, then merge across column-slice
    // blocks. Distances are clamped non-negative, so float bits are
    // order-isomorphic to uint: atomicMax/Min on the bit pattern.
    // No-positive slice: hps=2 -> dist=-1 -> clamp 0 -> no-op vs init 0.
    // No-negative slice: hns=-2 -> dist=3 -> no-op vs init 3.
    if (ln16 == 0) {
        #pragma unroll
        for (int r = 0; r < 8; ++r) {
            const int   irow = i0 + hi * 8 + r;
            const float hpd  = 1.0f - hps[r];   // hardest-positive distance
            const float hnd  = 1.0f - hns[r];   // hardest-negative distance
            atomicMax(&hpEnc[irow], __float_as_uint(fmaxf(hpd, 0.0f)));
            atomicMin(&hnEnc[irow], __float_as_uint(fminf(fmaxf(hnd, 0.0f), 3.0f)));
        }
    }
}

// --- per-row loss, wave-reduced, atomically accumulated --------------------
__global__ __launch_bounds__(256) void tl_rowloss(const int* __restrict__ labels,
                                                  const unsigned* __restrict__ hpEnc,
                                                  const unsigned* __restrict__ hnEnc,
                                                  const int* __restrict__ cc,
                                                  float* __restrict__ lossAcc,
                                                  int* __restrict__ cntAcc) {
    const int i = blockIdx.x * blockDim.x + threadIdx.x;
    float v = 0.0f;
    int   c = 0;
    if (i < N_EMB) {
        const int cnt = cc[labels[i]];
        if (cnt >= 2 && cnt < N_EMB) {       // exact any-pos & any-neg
            const float hp = __uint_as_float(hpEnc[i]);
            const float hn = __uint_as_float(hnEnc[i]);
            v = fmaxf(hp - hn + MARGIN, 0.0f);
            c = 1;
        }
    }
    #pragma unroll
    for (int off = 16; off >= 1; off >>= 1) {
        v += __shfl_xor(v, off, 32);
        c += __shfl_xor(c, off, 32);
    }
    if ((threadIdx.x & 31) == 0) {
        atomicAdd(lossAcc, v);
        atomicAdd(cntAcc, c);
    }
}

__global__ void tl_finalize(const float* __restrict__ lossAcc,
                            const int* __restrict__ cntAcc,
                            float* __restrict__ out) {
    int c = *cntAcc;
    out[0] = *lossAcc / (float)(c > 0 ? c : 1);
}

// ---------------------------------------------------------------------------
extern "C" void kernel_launch(void* const* d_in, const int* in_sizes, int n_in,
                              void* d_out, int out_size, void* d_ws, size_t ws_size,
                              hipStream_t stream) {
    const float* x      = (const float*)d_in[0];   // [8192,128] fp32
    const int*   labels = (const int*)d_in[1];     // [8192] int
    float*       out    = (float*)d_out;           // scalar

    // Workspace layout (~2.07 MB): f16 embeddings + mining state + accumulators
    char* ws = (char*)d_ws;
    _Float16* Eh      = (_Float16*)ws;                               // 2 MB
    unsigned* hpEnc   = (unsigned*)(ws + (size_t)N_EMB * D_EMB * 2); // 32 KB
    unsigned* hnEnc   = hpEnc + N_EMB;                               // 32 KB
    int*      cc      = (int*)(hnEnc + N_EMB);                       // 2 KB
    float*    lossAcc = (float*)(cc + N_CLS);
    int*      cntAcc  = (int*)(lossAcc + 1);

    tl_init     <<<N_EMB / 256, 256, 0, stream>>>(hpEnc, hnEnc, cc, lossAcc, cntAcc);
    tl_normalize<<<N_EMB / 8,   256, 0, stream>>>(x, Eh);
    tl_hist     <<<N_EMB / 256, 256, 0, stream>>>(labels, cc);
    tl_mine     <<<dim3(N_EMB / (16 * ROWTILES_PER_BLK), N_EMB / COLS_PER_SLICE),
                   256, 0, stream>>>(Eh, labels, hpEnc, hnEnc);
    tl_rowloss  <<<N_EMB / 256, 256, 0, stream>>>(labels, hpEnc, hnEnc, cc, lossAcc, cntAcc);
    tl_finalize <<<1, 1, 0, stream>>>(lossAcc, cntAcc, out);
}
